// MultiHeadSelfAttention_44238163149417
// MI455X (gfx1250) — matrix operand order repaired and checked by reading, compiled, not run on hardware
//
#include <hip/hip_runtime.h>

// ---------------------------------------------------------------------------
// MHA forward for MI455X (gfx1250): bf16 WMMA + TDM (tensor_load_to_lds).
// B=4, S=2048, D=1024, H=16, HD=64.
// ---------------------------------------------------------------------------

#define BATCH 4
#define SEQ   2048
#define DMODEL 1024
#define NHEAD 16
#define HDIM  64
#define MROWS (BATCH * SEQ)   // 8192

typedef __attribute__((ext_vector_type(16))) __bf16 v16bf;
typedef __attribute__((ext_vector_type(8)))  float  v8f;
typedef __attribute__((ext_vector_type(4)))  unsigned int v4u;
typedef __attribute__((ext_vector_type(8)))  int v8i;
typedef __attribute__((ext_vector_type(4)))  int v4i;

union AFrag {
    v16bf v;
    unsigned short s[16];
    unsigned int   u[8];
};
union BFragU {
    v16bf v;
    unsigned short s[16];
    uint4 q[2];
};

static __device__ inline unsigned short f2bf(float f) {
    union { float f; unsigned u; } v; v.f = f;
    unsigned r = v.u + 0x7FFFu + ((v.u >> 16) & 1u);   // round to nearest even
    return (unsigned short)(r >> 16);
}

static __device__ inline v8f wmma_bf16(v16bf a, v16bf b, v8f c) {
    return __builtin_amdgcn_wmma_f32_16x16x32_bf16(
        false, a, false, b, (short)0, c, false, false);
}

// B-fragment (32x16 bf16): per-lane contiguous 16 elems (k = 16g+e) at p.
static __device__ inline v16bf load_bfrag(const unsigned short* p) {
    BFragU b;
    b.q[0] = *(const uint4*)(p);
    b.q[1] = *(const uint4*)(p + 8);
    return b.v;
}

// A-fragment (16x32 bf16) from a 32-element row (16B-aligned).
// Contiguous k-pattern (k = 16g+e), matching load_bfrag:
// elems [0..7] at row+16g, elems [8..15] at row+16g+8.
static __device__ inline void load_afrag_row(AFrag& a, const unsigned short* row, int g) {
    const unsigned short* p = row + 16 * g;
    uint4 qa = *(const uint4*)(p);
    uint4 qb = *(const uint4*)(p + 8);
    a.u[0] = qa.x; a.u[1] = qa.y; a.u[2] = qa.z; a.u[3] = qa.w;
    a.u[4] = qb.x; a.u[5] = qb.y; a.u[6] = qb.z; a.u[7] = qb.w;
}

// LDS offset of a __shared__ pointer: flat-LDS aperture keeps the byte
// offset in addr[31:0] (ISA 10.2 aperture translation).
static __device__ inline unsigned lds_off(const void* p) {
    return (unsigned)(unsigned long long)p;
}

// ---------------------------------------------------------------------------
// TDM: 2D tile load global -> LDS (ISA 8.3/8.4 descriptor packing).
// data_size = 2 bytes; LDS row padding via pad_enable.
// ---------------------------------------------------------------------------
static __device__ inline void tdm_load_2d(unsigned ldsByteOff, const void* gaddr,
                                          unsigned tensor_d0, unsigned tensor_d1,
                                          unsigned tile_d0, unsigned tile_d1,
                                          unsigned long long stride0_elems,
                                          unsigned pad_interval_code,
                                          unsigned pad_amount_code) {
    unsigned long long ga = (unsigned long long)gaddr;
    v4u g0;
    g0[0] = 1u;                                            // count=1 (valid D#)
    g0[1] = ldsByteOff;                                    // lds_addr [63:32]
    g0[2] = (unsigned)(ga & 0xffffffffu);                  // global_addr[31:0]
    g0[3] = (unsigned)((ga >> 32) & 0x01ffffffu) | (2u << 30); // addr[56:32]|type=2
    v8i g1;
    g1[0] = (int)((1u << 16)                               // data_size = 2B
                | (1u << 20)                               // pad_enable
                | (pad_interval_code << 22)
                | (pad_amount_code << 25));
    g1[1] = (int)((tensor_d0 & 0xffffu) << 16);            // tensor_dim0 lo16
    g1[2] = (int)((tensor_d0 >> 16) | ((tensor_d1 & 0xffffu) << 16));
    g1[3] = (int)((tensor_d1 >> 16) | (tile_d0 << 16));    // dim1 hi | tile_dim0
    g1[4] = (int)(tile_d1 & 0xffffu);                      // tile_dim1 | tile_dim2=0
    g1[5] = (int)(unsigned)(stride0_elems & 0xffffffffu);  // dim0_stride[31:0]
    g1[6] = (int)(unsigned)(stride0_elems >> 32);          // dim0_stride[47:32]
    g1[7] = 0;
    v4i g2 = {0, 0, 0, 0};
    v4i g3 = {0, 0, 0, 0};
    v8i g4 = {0, 0, 0, 0, 0, 0, 0, 0};
    __builtin_amdgcn_tensor_load_to_lds(g0, g1, g2, g3, g4, 0);
}

// ---------------------------------------------------------------------------
// fp32 -> bf16 cast, 8 elements/thread (row-major, no transpose)
// ---------------------------------------------------------------------------
__global__ void cast_bf16_kernel(const float* __restrict__ src,
                                 unsigned short* __restrict__ dst, int n) {
    int i = (blockIdx.x * blockDim.x + threadIdx.x) * 8;
    if (i < n) {
        float4 f0 = *(const float4*)(src + i);
        float4 f1 = *(const float4*)(src + i + 4);
        union { uint4 q; unsigned short s[8]; } o;
        o.s[0] = f2bf(f0.x); o.s[1] = f2bf(f0.y);
        o.s[2] = f2bf(f0.z); o.s[3] = f2bf(f0.w);
        o.s[4] = f2bf(f1.x); o.s[5] = f2bf(f1.y);
        o.s[6] = f2bf(f1.z); o.s[7] = f2bf(f1.w);
        *(uint4*)(dst + i) = o.q;
    }
}

// ---------------------------------------------------------------------------
// fp32 [R][C] -> bf16 transposed [C][R] via 32x32 LDS tile (coalesced both ways)
// ---------------------------------------------------------------------------
__global__ void transpose_cast_kernel(const float* __restrict__ src,
                                      unsigned short* __restrict__ dst,
                                      int R, int C) {
    __shared__ unsigned short t[32][33];
    const int tx = threadIdx.x;       // 0..31
    const int ty = threadIdx.y;       // 0..7
    const int c0 = blockIdx.x * 32;
    const int r0 = blockIdx.y * 32;
#pragma unroll
    for (int i = 0; i < 4; ++i) {
        int r = ty + i * 8;
        t[tx][r] = f2bf(src[(size_t)(r0 + r) * C + c0 + tx]);
    }
    __syncthreads();
#pragma unroll
    for (int i = 0; i < 4; ++i) {
        int r = ty + i * 8;
        dst[(size_t)(c0 + r) * R + r0 + tx] = t[r][tx];
    }
}

// ---------------------------------------------------------------------------
// Tiled bf16 GEMM via TDM double-buffered staging.
//   C[M,N] = A[M,K] * Bt[N,K]^T + bias[N]      (Bt is pre-transposed weights)
//   mode 0: scatter bf16 into Qh[B,H,S,HD], Kh[B,H,S,HD], Vt[B,H,HD,S]
//   mode 1: store fp32 to Cout
// Block: 256 threads = 8 waves (4m x 2n), macro tile 64x128, BK=32.
// ---------------------------------------------------------------------------
#define BM 64
#define BN 128
#define BK 32
#define KP 40     // LDS row stride (elems): 32 + 4-dword TDM pad, 80B (16B mult)

__launch_bounds__(256)
__global__ void gemm_bf16_kernel(const unsigned short* __restrict__ A,
                                 const unsigned short* __restrict__ Bt,
                                 const float* __restrict__ bias,
                                 float* __restrict__ Cout,
                                 int M, int N, int K, int mode,
                                 unsigned short* __restrict__ Qh,
                                 unsigned short* __restrict__ Kh,
                                 unsigned short* __restrict__ Vt) {
    __shared__ unsigned short As[2][BM * KP];   // 2 x 5120 B
    __shared__ unsigned short Bs[2][BN * KP];   // 2 x 10240 B

    const int tid  = threadIdx.x;
    const int lane = tid & 31;
    const int wave = tid >> 5;
    const int wm = wave >> 1;        // 0..3
    const int wn = wave & 1;         // 0..1
    const int g  = lane >> 4;
    const int nn = lane & 15;

    const int row0 = blockIdx.y * BM;
    const int col0 = blockIdx.x * BN;

    const unsigned short* aTile = A  + (size_t)row0 * K;     // step +BK elems
    const unsigned short* bTile = Bt + (size_t)col0 * K;     // step +BK elems

    const int NT = K / BK;
    if (wave == 0) {
        tdm_load_2d(lds_off(&As[0][0]), aTile, (unsigned)K, (unsigned)M,
                    BK, BM, (unsigned long long)K, /*16 dw*/ 3, /*+4 dw*/ 3);
        tdm_load_2d(lds_off(&Bs[0][0]), bTile, (unsigned)K, (unsigned)N,
                    BK, BN, (unsigned long long)K, 3, 3);
    }
    __builtin_amdgcn_s_wait_tensorcnt(0);
    __syncthreads();

    v8f acc0 = 0, acc1 = 0, acc2 = 0, acc3 = 0;
    int cur = 0;
    for (int it = 0; it < NT; ++it) {
        if (it + 1 < NT && wave == 0) {
            const unsigned short* aN = aTile + (size_t)(it + 1) * BK;
            const unsigned short* bN = bTile + (size_t)(it + 1) * BK;
            tdm_load_2d(lds_off(&As[cur ^ 1][0]), aN, (unsigned)K, (unsigned)M,
                        BK, BM, (unsigned long long)K, 3, 3);
            tdm_load_2d(lds_off(&Bs[cur ^ 1][0]), bN, (unsigned)K, (unsigned)N,
                        BK, BN, (unsigned long long)K, 3, 3);
        }

        // batch all fragment loads, then the 4 WMMAs (graduated dscnt waits)
        AFrag a;
        load_afrag_row(a, &As[cur][(wm * 16 + nn) * KP], g);
        v16bf b0 = load_bfrag(&Bs[cur][(wn * 64 +  0 + nn) * KP + 16 * g]);
        v16bf b1 = load_bfrag(&Bs[cur][(wn * 64 + 16 + nn) * KP + 16 * g]);
        v16bf b2 = load_bfrag(&Bs[cur][(wn * 64 + 32 + nn) * KP + 16 * g]);
        v16bf b3 = load_bfrag(&Bs[cur][(wn * 64 + 48 + nn) * KP + 16 * g]);
        acc0 = wmma_bf16(a.v, b0, acc0);
        acc1 = wmma_bf16(a.v, b1, acc1);
        acc2 = wmma_bf16(a.v, b2, acc2);
        acc3 = wmma_bf16(a.v, b3, acc3);

        if (it + 1 < NT) {
            __builtin_amdgcn_s_wait_tensorcnt(0);
            __syncthreads();
            cur ^= 1;
        }
    }

    // ---- epilogue --------------------------------------------------------
    v8f accs[4] = {acc0, acc1, acc2, acc3};
#pragma unroll
    for (int t = 0; t < 4; ++t) {
        int c = col0 + wn * 64 + t * 16 + nn;
        float bv = bias[c];
        if (mode == 1) {
#pragma unroll
            for (int r = 0; r < 8; ++r) {
                int m = row0 + wm * 16 + r + 8 * g;
                Cout[(size_t)m * N + c] = accs[t][r] + bv;
            }
        } else {
            int which = c >> 10;        // 0=Q 1=K 2=V
            int rem   = c & 1023;
            int hh    = rem >> 6;
            int hd    = rem & 63;
#pragma unroll
            for (int r = 0; r < 8; ++r) {
                int m  = row0 + wm * 16 + r + 8 * g;
                int bb = m >> 11;
                int ss = m & 2047;
                unsigned short bf = f2bf(accs[t][r] + bv);
                size_t bhh = (size_t)((bb << 4) + hh);
                if (which == 0)
                    Qh[(bhh * SEQ + ss) * HDIM + hd] = bf;
                else if (which == 1)
                    Kh[(bhh * SEQ + ss) * HDIM + hd] = bf;
                else
                    Vt[(bhh * HDIM + hd) * SEQ + ss] = bf;
            }
        }
    }
}

// ---------------------------------------------------------------------------
// Flash attention: one wave = one 16-query tile of one (b,h).
// ---------------------------------------------------------------------------
#define AW 4   // waves per block
#define PLP 40 // P LDS row stride (elems), 80B keeps b128 alignment

__launch_bounds__(128)
__global__ void attn_kernel(const unsigned short* __restrict__ Qh,
                            const unsigned short* __restrict__ Kh,
                            const unsigned short* __restrict__ Vt,
                            unsigned short* __restrict__ ctx) {
    __shared__ unsigned short Pl[AW][16 * PLP];

    const int tid  = threadIdx.x;
    const int lane = tid & 31;
    const int w    = tid >> 5;
    const int g    = lane >> 4;
    const int nn   = lane & 15;

    const int flat = blockIdx.x * AW + w;       // 0 .. B*H*(S/16)-1
    const int qt   = flat & 127;                // query tile (S/16 = 128)
    const int bh   = flat >> 7;
    const int h    = bh & 15;
    const int b    = bh >> 4;

    const unsigned short* Qp = Qh + (size_t)bh * SEQ * HDIM;
    const unsigned short* Kp = Kh + (size_t)bh * SEQ * HDIM;
    const unsigned short* Vp = Vt + (size_t)bh * HDIM * SEQ;

    // Q fragments (k 0..31, 32..63)
    AFrag a0, a1;
    {
        const unsigned short* qrow = Qp + (size_t)(qt * 16 + nn) * HDIM;
        load_afrag_row(a0, qrow, g);
        load_afrag_row(a1, qrow + 32, g);
    }

    v8f O0 = 0, O1 = 0, O2 = 0, O3 = 0;
    float mrow[8], lrow[8];
#pragma unroll
    for (int r = 0; r < 8; ++r) { mrow[r] = -__builtin_inff(); lrow[r] = 0.f; }

    const int kend = qt * 16 + 16;
    for (int kb = 0; kb < kend; kb += 32) {
        // batched K-fragment loads, then the 4 score WMMAs
        const unsigned short* kRow0 = Kp + (size_t)(kb + nn) * HDIM + 16 * g;
        const unsigned short* kRow1 = Kp + (size_t)(kb + 16 + nn) * HDIM + 16 * g;
        v16bf k00 = load_bfrag(kRow0);
        v16bf k01 = load_bfrag(kRow0 + 32);
        v16bf k10 = load_bfrag(kRow1);
        v16bf k11 = load_bfrag(kRow1 + 32);
        v8f s0 = 0, s1 = 0;
        s0 = wmma_bf16(a0.v, k00, s0);
        s0 = wmma_bf16(a1.v, k01, s0);
        s1 = wmma_bf16(a0.v, k10, s1);
        s1 = wmma_bf16(a1.v, k11, s1);

        // V fragments issued early: global latency hides under the softmax
        const unsigned short* vCol = Vp + (size_t)nn * SEQ + kb + 16 * g;
        v16bf v0 = load_bfrag(vCol);
        v16bf v1 = load_bfrag(vCol + (size_t)16 * SEQ);
        v16bf v2 = load_bfrag(vCol + (size_t)32 * SEQ);
        v16bf v3 = load_bfrag(vCol + (size_t)48 * SEQ);

        const bool domask = (kb + 31 > qt * 16);
        float al[8];
#pragma unroll
        for (int r = 0; r < 8; ++r) {
            float x0 = s0[r] * 0.125f;
            float x1 = s1[r] * 0.125f;
            if (domask) {
                int q = qt * 16 + r + 8 * g;
                if (kb + nn > q)      x0 = -__builtin_inff();
                if (kb + 16 + nn > q) x1 = -__builtin_inff();
            }
            float mx = fmaxf(x0, x1);
            mx = fmaxf(mx, __shfl_xor(mx, 1, 32));
            mx = fmaxf(mx, __shfl_xor(mx, 2, 32));
            mx = fmaxf(mx, __shfl_xor(mx, 4, 32));
            mx = fmaxf(mx, __shfl_xor(mx, 8, 32));
            float mnew = fmaxf(mrow[r], mx);
            float alpha = __expf(mrow[r] - mnew);
            float p0 = __expf(x0 - mnew);
            float p1 = __expf(x1 - mnew);
            float rs = p0 + p1;
            rs += __shfl_xor(rs, 1, 32);
            rs += __shfl_xor(rs, 2, 32);
            rs += __shfl_xor(rs, 4, 32);
            rs += __shfl_xor(rs, 8, 32);
            lrow[r] = lrow[r] * alpha + rs;
            mrow[r] = mnew;
            al[r]   = alpha;
            int m = r + 8 * g;
            Pl[w][m * PLP + nn]      = f2bf(p0);
            Pl[w][m * PLP + nn + 16] = f2bf(p1);
        }
#pragma unroll
        for (int r = 0; r < 8; ++r) {
            O0[r] *= al[r]; O1[r] *= al[r]; O2[r] *= al[r]; O3[r] *= al[r];
        }

        // same-wave LDS RAW (DS ops are in-order per wave)
        asm volatile("s_wait_dscnt 0" ::: "memory");

        AFrag pa;
        load_afrag_row(pa, &Pl[w][nn * PLP], g);

        O0 = wmma_bf16(pa.v, v0, O0);
        O1 = wmma_bf16(pa.v, v1, O1);
        O2 = wmma_bf16(pa.v, v2, O2);
        O3 = wmma_bf16(pa.v, v3, O3);
    }

#pragma unroll
    for (int r = 0; r < 8; ++r) {
        float inv = 1.0f / lrow[r];
        int q = qt * 16 + r + 8 * g;
        size_t base = ((size_t)b * SEQ + q) * DMODEL + h * HDIM;
        ctx[base +  0 + nn] = f2bf(O0[r] * inv);
        ctx[base + 16 + nn] = f2bf(O1[r] * inv);
        ctx[base + 32 + nn] = f2bf(O2[r] * inv);
        ctx[base + 48 + nn] = f2bf(O3[r] * inv);
    }
}

// ---------------------------------------------------------------------------
// host launcher
// ---------------------------------------------------------------------------
extern "C" void kernel_launch(void* const* d_in, const int* in_sizes, int n_in,
                              void* d_out, int out_size, void* d_ws, size_t ws_size,
                              hipStream_t stream) {
    const float* x    = (const float*)d_in[0];
    // d_in[1] = causal mask (computed analytically, unused)
    const float* Wqkv = (const float*)d_in[2];
    const float* bqkv = (const float*)d_in[3];
    const float* Wout = (const float*)d_in[4];
    const float* bout = (const float*)d_in[5];

    char* ws = (char*)d_ws;
    size_t off = 0;
    unsigned short* xbf    = (unsigned short*)(ws + off); off += (size_t)MROWS * DMODEL * 2;
    unsigned short* wqkvT  = (unsigned short*)(ws + off); off += (size_t)DMODEL * 3 * DMODEL * 2;
    unsigned short* woutT  = (unsigned short*)(ws + off); off += (size_t)DMODEL * DMODEL * 2;
    unsigned short* Qh     = (unsigned short*)(ws + off); off += (size_t)MROWS * DMODEL * 2;
    unsigned short* Kh     = (unsigned short*)(ws + off); off += (size_t)MROWS * DMODEL * 2;
    unsigned short* Vt     = (unsigned short*)(ws + off); off += (size_t)MROWS * DMODEL * 2;
    unsigned short* ctx    = (unsigned short*)(ws + off); off += (size_t)MROWS * DMODEL * 2;

    const int nx = MROWS * DMODEL;
    cast_bf16_kernel<<<(nx / 8 + 255) / 256, 256, 0, stream>>>(x, xbf, nx);
    // Wqkv [1024][3072] -> wqkvT [3072][1024] bf16
    transpose_cast_kernel<<<dim3(3 * DMODEL / 32, DMODEL / 32), dim3(32, 8), 0, stream>>>(
        Wqkv, wqkvT, DMODEL, 3 * DMODEL);
    // Wout [1024][1024] -> woutT [1024][1024] bf16
    transpose_cast_kernel<<<dim3(DMODEL / 32, DMODEL / 32), dim3(32, 8), 0, stream>>>(
        Wout, woutT, DMODEL, DMODEL);

    // QKV projection + head scatter: [8192,1024] x [1024,3072]
    gemm_bf16_kernel<<<dim3(3 * DMODEL / BN, MROWS / BM), 256, 0, stream>>>(
        xbf, wqkvT, bqkv, nullptr, MROWS, 3 * DMODEL, DMODEL, 0, Qh, Kh, Vt);

    // flash attention: B*H*(S/16) = 8192 waves, 4 per block
    attn_kernel<<<(BATCH * NHEAD * (SEQ / 16)) / AW, 32 * AW, 0, stream>>>(
        Qh, Kh, Vt, ctx);

    // output projection: [8192,1024] x [1024,1024] -> fp32 d_out
    gemm_bf16_kernel<<<dim3(DMODEL / BN, MROWS / BM), 256, 0, stream>>>(
        ctx, woutT, bout, (float*)d_out, MROWS, DMODEL, DMODEL, 1,
        nullptr, nullptr, nullptr);
}
